// GraphUnet_27590869909518
// MI455X (gfx1250) — matrix operand order repaired and checked by reading, not compiled, not run on hardware
//
#include <hip/hip_runtime.h>
#include <hip/hip_bf16.h>

// ---------------------------------------------------------------------------
// Graph U-Net forward for MI455X (gfx1250, wave32, WMMA).
// Heavy op: H = (A o (M^T M / R)) @ X  -> stream fp32 A ONCE per layer,
// on-the-fly mask scale, bf16 fragments staged in LDS in WMMA layout,
// v_wmma_f32_16x16x32_bf16 with NFT register accumulators per wave (A reuse).
// All per-thread loop-invariant addressing hoisted out of the K loop.
// ---------------------------------------------------------------------------

typedef __attribute__((ext_vector_type(16))) __bf16 v16bf;
typedef __attribute__((ext_vector_type(8)))  float  v8f;

#define N0 8192
#define FIN 256
#define HD 128
#define NL1 6553   /* int(0.8*8192) */
#define NL2 3931   /* int(0.6*6553) */
#define RREP 5
#define TA 64      /* rows per block */
#define TK 32      /* K chunk        */

__device__ __forceinline__ unsigned int hashu(unsigned int x) {
    x ^= x >> 16; x *= 0x7feb352du;
    x ^= x >> 15; x *= 0x846ca68bu;
    x ^= x >> 16; return x;
}

// A-fragment position for element (mrow 0..15, k 0..31):
//   lane = mrow + 16*(k>>4),      e = k&15
// B-fragment position for element (k 0..31, ncol 0..15):
//   lane = ncol + 16*(k>>4),      e = k&15
// C tile: VGPR v -> row = v + 8*(lane>=16), col = lane&15.

__global__ void make_keep(unsigned char* keep, int n, unsigned int seed) {
    int i = blockIdx.x * blockDim.x + threadIdx.x;
    if (i >= n) return;
    unsigned char b = 0;
    #pragma unroll
    for (int r = 0; r < RREP; ++r) {
        unsigned int h = hashu(hashu(seed ^ (0x9e3779b9u * (r + 1))) ^ (unsigned)i);
        if ((h & 0xFFFFu) >= 19661u) b |= (1u << r);   // keep w.p. ~0.7
    }
    keep[i] = b;
}

__global__ void iota_kernel(int* p, int n) {
    int i = blockIdx.x * blockDim.x + threadIdx.x;
    if (i < n) p[i] = i;
}

// dst[r*dstld + dstoff + c] = (bf16) src[r*cols + c]
__global__ void cvt_bf16(const float* __restrict__ src, __bf16* __restrict__ dst,
                         int rows, int cols, int dstld, int dstoff) {
    int e = blockIdx.x * blockDim.x + threadIdx.x;
    if (e >= rows * cols) return;
    int r = e / cols, c = e - r * cols;
    dst[(long)r * dstld + dstoff + c] = (__bf16)src[(long)r * cols + c];
}

__global__ void fill_zero(float* p, long n) {
    long i = (long)blockIdx.x * blockDim.x + threadIdx.x;
    if (i < n) p[i] = 0.f;
}

__global__ void scatter_rows(const float* __restrict__ src, const int* __restrict__ idx,
                             float* __restrict__ dst, int mrows, int cols) {
    int e = blockIdx.x * blockDim.x + threadIdx.x;
    if (e >= mrows * cols) return;
    int r = e / cols, c = e - r * cols;
    dst[(long)idx[r] * cols + c] = src[(long)r * cols + c];
}

__global__ void gather_scale(const float* __restrict__ src, const int* __restrict__ idx,
                             const float* __restrict__ vals, float* __restrict__ dst,
                             int mrows, int cols) {
    int e = blockIdx.x * blockDim.x + threadIdx.x;
    if (e >= mrows * cols) return;
    int r = e / cols, c = e - r * cols;
    dst[(long)r * cols + c] = src[(long)idx[r] * cols + c] * vals[r];
}

__global__ void compose_idx(int* __restrict__ gout, const int* __restrict__ outer,
                            const int* __restrict__ inner, int n) {
    int i = blockIdx.x * blockDim.x + threadIdx.x;
    if (i < n) gout[i] = outer[inner[i]];
}

__global__ void pool_scores(const float* __restrict__ X, const float* __restrict__ Wp,
                            const float* __restrict__ bp, float* __restrict__ scores, int m) {
    int i = blockIdx.x * blockDim.x + threadIdx.x;
    if (i >= m) return;
    float s = bp[0];
    for (int k = 0; k < HD; ++k) s += X[(long)i * HD + k] * Wp[k];
    scores[i] = 1.f / (1.f + __expf(-s * 0.01f));
}

// Exact descending top-k by rank counting (ties broken by index).
__global__ void topk_select(const float* __restrict__ scores, int n, int k,
                            int* __restrict__ idx, float* __restrict__ vals) {
    int i = blockIdx.x;
    __shared__ int cnt;
    if (threadIdx.x == 0) cnt = 0;
    __syncthreads();
    float si = scores[i];
    int local = 0;
    for (int j = threadIdx.x; j < n; j += blockDim.x) {
        float sj = scores[j];
        if (sj > si || (sj == si && j < i)) ++local;
    }
    atomicAdd(&cnt, local);
    __syncthreads();
    if (threadIdx.x == 0 && cnt < k) { idx[cnt] = i; vals[cnt] = si; }
}

// ---------------------------------------------------------------------------
// H[m,F] = (A o (M^T M/R))[gidx][:,gidx] @ Xb   (Xb local [m,F] bf16)
// One block = 64 rows x ALL F features; A streamed from HBM exactly once.
// 8 waves: wr = wave>>1 row-tile, feature tiles ft = (wave&1) + 2*t.
// ---------------------------------------------------------------------------
template <int NFT>
__global__ __launch_bounds__(256, 1)
void gcn_spmm(const float* __restrict__ A, long lda,
              const int* __restrict__ gidx,            // always valid (iota for full)
              const unsigned char* __restrict__ keep,  // local keep bytes [m]
              const __bf16* __restrict__ Xb,
              float* __restrict__ Hout, int m) {
    constexpr int F = NFT * 32;
    constexpr int FSH = (F == 256) ? 8 : 7;
    __shared__ __align__(32) __bf16 aFrag[4][32][16];
    __shared__ __align__(32) __bf16 bFrag[2 * NFT][32][16];
    __shared__ int sgrow[TA];
    __shared__ unsigned char skeepR[TA];

    const int r0 = blockIdx.x * TA;
    const int tid = threadIdx.x;
    const int wave = tid >> 5, lane = tid & 31;
    const int wr = wave >> 1, wc = wave & 1;
    const int l15 = lane & 15;

    if (tid < TA) {
        int r = r0 + tid;
        sgrow[tid]  = (r < m) ? gidx[r] : 0;
        skeepR[tid] = (r < m) ? keep[r] : 0;
    }
    __syncthreads();

    // Loop-invariant per-thread staging state (hoisted into registers):
    // column slot ak = tid&31, rows ar = (tid>>5) + 8i.
    const int ak  = tid & 31;
    const int ar0 = tid >> 5;
    const int at  = ak & 15;                       // e within A fragment
    const int ahl = (ak >> 4) << 4;                // lane-half offset

    const float* rowp[8];
    float        kscl[8];      // promoted keep byte (as float-scaled later)
    unsigned     krow[8];
    bool         rowok[8];
    __bf16*      aslot[8];
    #pragma unroll
    for (int i = 0; i < 8; ++i) {
        int ar   = ar0 + (i << 3);
        rowp[i]  = A + (long)sgrow[ar] * lda;
        krow[i]  = skeepR[ar];
        rowok[i] = (r0 + ar) < m;
        aslot[i] = &aFrag[ar >> 4][(ar & 15) + ahl][at];
        kscl[i]  = 0.f; (void)kscl;
    }
    const float* prow = A + (long)sgrow[tid & (TA - 1)] * lda;  // prefetch row

    v8f acc[NFT];
    #pragma unroll
    for (int t = 0; t < NFT; ++t) acc[t] = (v8f){};

    auto compute = [&]() {
        v16bf avec = *(const v16bf*)&aFrag[wr][lane][0];
        v16bf bcur = *(const v16bf*)&bFrag[wc][lane][0];
        #pragma unroll
        for (int t = 0; t < NFT; ++t) {
            v16bf bnext;
            if (t + 1 < NFT) bnext = *(const v16bf*)&bFrag[wc + 2 * (t + 1)][lane][0];
            acc[t] = __builtin_amdgcn_wmma_f32_16x16x32_bf16(
                false, avec, false, bcur, (short)0, acc[t], false, false);
            if (t + 1 < NFT) bcur = bnext;
        }
    };

    const bool rowfull = (r0 + TA <= m);
    const int  kmain   = rowfull ? (m & ~(TK - 1)) : 0;

    int kk = 0;
    for (; kk < kmain; kk += TK) {          // branch-free interior path
        __syncthreads();
        {
            const int gc = gidx[kk + ak];
            const unsigned kc = keep[kk + ak];
            #pragma unroll
            for (int i = 0; i < 8; ++i) {
                float v = rowp[i][gc] *
                          ((float)__builtin_popcount(krow[i] & kc) * 0.2f);
                *aslot[i] = (__bf16)v;
            }
            const __bf16* xsrc = Xb + (long)kk * F;   // 32 contiguous X rows
            #pragma unroll
            for (int q = tid; q < (TK << FSH); q += 256) {
                int k = q >> FSH, n = q & (F - 1);
                bFrag[n >> 4][(n & 15) + ((k >> 4) << 4)][k & 15] = xsrc[q];
            }
            if (tid < TA && kk + TK < m) {  // prefetch next chunk row heads
                const float* p = prow + gidx[kk + TK];
                __builtin_prefetch(p, 0, 0);
                __builtin_prefetch(p + 16, 0, 0);
            }
        }
        __syncthreads();
        compute();
    }
    for (; kk < m; kk += TK) {              // guarded tail / edge blocks
        __syncthreads();
        {
            const bool kok = (kk + ak) < m;
            const int gc = kok ? gidx[kk + ak] : 0;
            const unsigned kc = kok ? keep[kk + ak] : 0u;
            #pragma unroll
            for (int i = 0; i < 8; ++i) {
                float v = 0.f;
                if (kok && rowok[i])
                    v = rowp[i][gc] *
                        ((float)__builtin_popcount(krow[i] & kc) * 0.2f);
                *aslot[i] = (__bf16)v;
            }
            const __bf16* xsrc = Xb + (long)kk * F;
            #pragma unroll
            for (int q = tid; q < (TK << FSH); q += 256) {
                int k = q >> FSH, n = q & (F - 1);
                __bf16 v = (__bf16)0.f;
                if (kk + k < m) v = xsrc[q];
                bFrag[n >> 4][(n & 15) + ((k >> 4) << 4)][k & 15] = v;
            }
        }
        __syncthreads();
        compute();
    }

    const int rbase = r0 + wr * 16 + ((lane >> 4) << 3);
    if (rowfull) {
        #pragma unroll
        for (int t = 0; t < NFT; ++t) {
            int col = (wc + 2 * t) * 16 + l15;
            #pragma unroll
            for (int v = 0; v < 8; ++v)
                Hout[(long)(rbase + v) * F + col] = acc[t][v];
        }
    } else {
        #pragma unroll
        for (int t = 0; t < NFT; ++t) {
            int col = (wc + 2 * t) * 16 + l15;
            #pragma unroll
            for (int v = 0; v < 8; ++v)
                if (rbase + v < m) Hout[(long)(rbase + v) * F + col] = acc[t][v];
        }
    }
}

// ---------------------------------------------------------------------------
// Y[m,N] = relu(Hin[m,K] @ W[N,K]^T + b) (+ residual).  Same frag layout;
// block = 64 rows x ALL N outputs.  N templated, K runtime (multiple of 32).
// ---------------------------------------------------------------------------
template <int NNT>
__global__ __launch_bounds__(256, 1)
void linear_relu(const float* __restrict__ Hin, int K,
                 const float* __restrict__ W, const float* __restrict__ b,
                 const float* __restrict__ residual,
                 float* __restrict__ Y, int m) {
    constexpr int N = NNT * 32;
    __shared__ __align__(32) __bf16 aFrag[4][32][16];
    __shared__ __align__(32) __bf16 bFrag[2 * NNT][32][16];

    const int r0 = blockIdx.x * TA;
    const int tid = threadIdx.x;
    const int wave = tid >> 5, lane = tid & 31;
    const int wr = wave >> 1, wc = wave & 1;
    const int l15 = lane & 15;

    const int ak  = tid & 31;
    const int ar0 = tid >> 5;
    const int at  = ak & 15;
    const int ahl = (ak >> 4) << 4;

    const float* rowp[8];
    bool         rowok[8];
    __bf16*      aslot[8];
    #pragma unroll
    for (int i = 0; i < 8; ++i) {
        int ar   = ar0 + (i << 3);
        rowok[i] = (r0 + ar) < m;
        rowp[i]  = Hin + (long)(rowok[i] ? (r0 + ar) : 0) * K + ak;
        aslot[i] = &aFrag[ar >> 4][(ar & 15) + ahl][at];
    }

    // W slot per q is kk-invariant: k = q>>NSH_dyn? N is constexpr here.
    constexpr int NSH = (N == 256) ? 8 : 7;

    v8f acc[NNT];
    #pragma unroll
    for (int t = 0; t < NNT; ++t) acc[t] = (v8f){};

    const bool rowfull = (r0 + TA <= m);

    for (int kk = 0; kk < K; kk += TK) {
        __syncthreads();
        if (rowfull) {
            #pragma unroll
            for (int i = 0; i < 8; ++i)
                *aslot[i] = (__bf16)rowp[i][kk];
        } else {
            #pragma unroll
            for (int i = 0; i < 8; ++i) {
                float v = rowok[i] ? rowp[i][kk] : 0.f;
                *aslot[i] = (__bf16)v;
            }
        }
        #pragma unroll
        for (int q = tid; q < (TK << NSH); q += 256) {
            int k = q >> NSH, n = q & (N - 1);
            bFrag[n >> 4][(n & 15) + ((k >> 4) << 4)][k & 15] =
                (__bf16)W[(long)n * K + kk + k];
        }
        __syncthreads();

        v16bf avec = *(const v16bf*)&aFrag[wr][lane][0];
        v16bf bcur = *(const v16bf*)&bFrag[wc][lane][0];
        #pragma unroll
        for (int t = 0; t < NNT; ++t) {
            v16bf bnext;
            if (t + 1 < NNT) bnext = *(const v16bf*)&bFrag[wc + 2 * (t + 1)][lane][0];
            acc[t] = __builtin_amdgcn_wmma_f32_16x16x32_bf16(
                false, avec, false, bcur, (short)0, acc[t], false, false);
            if (t + 1 < NNT) bcur = bnext;
        }
    }

    const int rbase = r0 + wr * 16 + ((lane >> 4) << 3);
    #pragma unroll
    for (int t = 0; t < NNT; ++t) {
        int col = (wc + 2 * t) * 16 + l15;
        #pragma unroll
        for (int v = 0; v < 8; ++v) {
            int row = rbase + v;
            if (row < m) {
                float y = acc[t][v] + b[col];
                y = y > 0.f ? y : 0.f;
                if (residual) y += residual[(long)row * N + col];
                Y[(long)row * N + col] = y;
            }
        }
    }
}

// ---------------------------------------------------------------------------
extern "C" void kernel_launch(void* const* d_in, const int* in_sizes, int n_in,
                              void* d_out, int out_size, void* d_ws, size_t ws_size,
                              hipStream_t stream) {
    const float* A       = (const float*)d_in[0];
    const float* Xin     = (const float*)d_in[1];
    const float* W_start = (const float*)d_in[2];
    const float* b_start = (const float*)d_in[3];
    const float* W_down0 = (const float*)d_in[4];
    const float* b_down0 = (const float*)d_in[5];
    const float* W_down1 = (const float*)d_in[6];
    const float* b_down1 = (const float*)d_in[7];
    const float* W_bot   = (const float*)d_in[8];
    const float* b_bot   = (const float*)d_in[9];
    const float* W_up0   = (const float*)d_in[10];
    const float* b_up0   = (const float*)d_in[11];
    const float* W_up1   = (const float*)d_in[12];
    const float* b_up1   = (const float*)d_in[13];
    const float* W_end   = (const float*)d_in[14];
    const float* b_end   = (const float*)d_in[15];
    const float* Wp0     = (const float*)d_in[16];
    const float* bp0     = (const float*)d_in[17];
    const float* Wp1     = (const float*)d_in[18];
    const float* bp1     = (const float*)d_in[19];

    float* Xfinal = (float*)d_out;                 // [8192,256]
    float* X0     = (float*)d_out + (long)N0*FIN;  // start_gcn_outs / org_X [8192,128]

    char* w = (char*)d_ws;
    auto carve = [&](size_t bytes) { char* p = w; w += (bytes + 255) & ~(size_t)255; return p; };
    unsigned char* keep = (unsigned char*)carve(N0);
    int*   iota  = (int*)carve((size_t)N0 * 4);
    __bf16* Xb   = (__bf16*)carve((size_t)N0 * FIN * 2);
    float* Hbuf  = (float*)carve((size_t)N0 * FIN * 4);
    float* X1    = (float*)carve((size_t)N0 * HD * 4);
    float* X1p   = (float*)carve((size_t)NL1 * HD * 4);
    float* X2    = (float*)carve((size_t)NL1 * HD * 4);
    float* X2p   = (float*)carve((size_t)NL2 * HD * 4);
    float* X3    = (float*)carve((size_t)NL2 * HD * 4);
    float* scat1 = (float*)carve((size_t)NL1 * HD * 4);
    float* X4    = (float*)carve((size_t)NL1 * HD * 4);
    float* scat0 = (float*)carve((size_t)N0 * HD * 4);
    float* X5    = (float*)carve((size_t)N0 * HD * 4);
    float* scores= (float*)carve((size_t)N0 * 4);
    int*   idx0  = (int*)carve((size_t)NL1 * 4);
    float* vals0 = (float*)carve((size_t)NL1 * 4);
    int*   idx1  = (int*)carve((size_t)NL2 * 4);
    float* vals1 = (float*)carve((size_t)NL2 * 4);
    int*   gidx1 = (int*)carve((size_t)NL2 * 4);

    auto blks = [](long n) { return (unsigned)((n + 255) / 256); };
    auto rt   = [](int m) { return (unsigned)((m + TA - 1) / TA); };

    iota_kernel<<<blks(N0), 256, 0, stream>>>(iota, N0);

    // ---- start layer: X0 = drop_gcn(A, X, W_start) ----
    make_keep<<<blks(N0), 256, 0, stream>>>(keep, N0, 0xC0FFEE00u);
    cvt_bf16<<<blks((long)N0*FIN), 256, 0, stream>>>(Xin, Xb, N0, FIN, FIN, 0);
    gcn_spmm<8><<<rt(N0), 256, 0, stream>>>(A, N0, iota, keep, Xb, Hbuf, N0);
    linear_relu<4><<<rt(N0), 256, 0, stream>>>(Hbuf, FIN, W_start, b_start, nullptr, X0, N0);

    // ---- down0 ----
    make_keep<<<blks(N0), 256, 0, stream>>>(keep, N0, 0xC0FFEE01u);
    cvt_bf16<<<blks((long)N0*HD), 256, 0, stream>>>(X0, Xb, N0, HD, HD, 0);
    gcn_spmm<4><<<rt(N0), 256, 0, stream>>>(A, N0, iota, keep, Xb, Hbuf, N0);
    linear_relu<4><<<rt(N0), 256, 0, stream>>>(Hbuf, HD, W_down0, b_down0, nullptr, X1, N0);
    // pool0
    pool_scores<<<blks(N0), 256, 0, stream>>>(X1, Wp0, bp0, scores, N0);
    topk_select<<<N0, 256, 0, stream>>>(scores, N0, NL1, idx0, vals0);
    gather_scale<<<blks((long)NL1*HD), 256, 0, stream>>>(X1, idx0, vals0, X1p, NL1, HD);

    // ---- down1 ----
    make_keep<<<blks(NL1), 256, 0, stream>>>(keep, NL1, 0xC0FFEE02u);
    cvt_bf16<<<blks((long)NL1*HD), 256, 0, stream>>>(X1p, Xb, NL1, HD, HD, 0);
    gcn_spmm<4><<<rt(NL1), 256, 0, stream>>>(A, N0, idx0, keep, Xb, Hbuf, NL1);
    linear_relu<4><<<rt(NL1), 256, 0, stream>>>(Hbuf, HD, W_down1, b_down1, nullptr, X2, NL1);
    // pool1
    pool_scores<<<blks(NL1), 256, 0, stream>>>(X2, Wp1, bp1, scores, NL1);
    topk_select<<<NL1, 256, 0, stream>>>(scores, NL1, NL2, idx1, vals1);
    gather_scale<<<blks((long)NL2*HD), 256, 0, stream>>>(X2, idx1, vals1, X2p, NL2, HD);
    compose_idx<<<blks(NL2), 256, 0, stream>>>(gidx1, idx0, idx1, NL2);

    // ---- bottom ----
    make_keep<<<blks(NL2), 256, 0, stream>>>(keep, NL2, 0xC0FFEE03u);
    cvt_bf16<<<blks((long)NL2*HD), 256, 0, stream>>>(X2p, Xb, NL2, HD, HD, 0);
    gcn_spmm<4><<<rt(NL2), 256, 0, stream>>>(A, N0, gidx1, keep, Xb, Hbuf, NL2);
    linear_relu<4><<<rt(NL2), 256, 0, stream>>>(Hbuf, HD, W_bot, b_bot, nullptr, X3, NL2);

    // ---- up0 (level 1): unpool idx1 into NL1, gcn, + skip X2 ----
    fill_zero<<<blks((long)NL1*HD), 256, 0, stream>>>(scat1, (long)NL1*HD);
    scatter_rows<<<blks((long)NL2*HD), 256, 0, stream>>>(X3, idx1, scat1, NL2, HD);
    make_keep<<<blks(NL1), 256, 0, stream>>>(keep, NL1, 0xC0FFEE04u);
    cvt_bf16<<<blks((long)NL1*HD), 256, 0, stream>>>(scat1, Xb, NL1, HD, HD, 0);
    gcn_spmm<4><<<rt(NL1), 256, 0, stream>>>(A, N0, idx0, keep, Xb, Hbuf, NL1);
    linear_relu<4><<<rt(NL1), 256, 0, stream>>>(Hbuf, HD, W_up0, b_up0, X2, X4, NL1);

    // ---- up1 (level 0): unpool idx0 into N0, gcn, + skip X1 ----
    fill_zero<<<blks((long)N0*HD), 256, 0, stream>>>(scat0, (long)N0*HD);
    scatter_rows<<<blks((long)NL1*HD), 256, 0, stream>>>(X4, idx0, scat0, NL1, HD);
    make_keep<<<blks(N0), 256, 0, stream>>>(keep, N0, 0xC0FFEE05u);
    cvt_bf16<<<blks((long)N0*HD), 256, 0, stream>>>(scat0, Xb, N0, HD, HD, 0);
    gcn_spmm<4><<<rt(N0), 256, 0, stream>>>(A, N0, iota, keep, Xb, Hbuf, N0);
    linear_relu<4><<<rt(N0), 256, 0, stream>>>(Hbuf, HD, W_up1, b_up1, X1, X5, N0);

    // ---- end layer: concat [X5, org_X=X0] -> drop_gcn with W_end ----
    cvt_bf16<<<blks((long)N0*HD), 256, 0, stream>>>(X5, Xb, N0, HD, FIN, 0);
    cvt_bf16<<<blks((long)N0*HD), 256, 0, stream>>>(X0, Xb, N0, HD, FIN, HD);
    make_keep<<<blks(N0), 256, 0, stream>>>(keep, N0, 0xC0FFEE06u);
    gcn_spmm<8><<<rt(N0), 256, 0, stream>>>(A, N0, iota, keep, Xb, Hbuf, N0);
    linear_relu<8><<<rt(N0), 256, 0, stream>>>(Hbuf, FIN, W_end, b_end, nullptr, Xfinal, N0);
}
